// RegionProposalNetwork_9852654977110
// MI455X (gfx1250) — matrix operand order repaired and checked by reading, not compiled, not run on hardware
//
#include <hip/hip_runtime.h>
#include <math.h>

typedef __attribute__((ext_vector_type(16))) __bf16        v16bf;
typedef __attribute__((ext_vector_type(8)))  float         v8f;
typedef __attribute__((ext_vector_type(4)))  unsigned int  u32x4;

#define CC    256
#define NIMG  2
#define NLVL  5
#define KSEL  4450          // sum of min(1000, H*W*3) over levels
#define TOTA  153450        // total anchors per image
#define NP2   16384         // sort pad (>= NIMG*KSEL = 8900)
#define NEGINF (-3.0e38f)

struct LvlInfo { int aoff[NLVL]; int nA[NLVL]; int koff[NLVL]; int kcnt[NLVL]; };

__device__ __forceinline__ unsigned short f2bf(float f) {
  union { float f; unsigned u; } x; x.f = f;
  unsigned r = x.u + 0x7fffu + ((x.u >> 16) & 1u);   // RNE
  return (unsigned short)(r >> 16);
}
__device__ __forceinline__ float bf2f(unsigned short h) {
  union { unsigned u; float f; } x; x.u = ((unsigned)h) << 16;
  return x.f;
}

// ---------------------------------------------------------------------------
// Pack conv weights (O=256, I=256, 3, 3) fp32 -> bf16 WMMA B fragments.
// Layout: [kc=72][octile=16][lane=32][e=16] with K = (lane>=16 ? 16 : 0) + e,
// N = lane&15, k = kc*32 + K, tap = k>>8, ic = k&255 (k ordering: tap-major).
// ---------------------------------------------------------------------------
__global__ void pack_weights(const float* __restrict__ w, unsigned short* __restrict__ wfrag) {
  int idx = blockIdx.x * blockDim.x + threadIdx.x;
  if (idx >= 72 * 16 * 32 * 16) return;
  int e    =  idx        & 15;
  int lane = (idx >> 4)  & 31;
  int ot   = (idx >> 9)  & 15;
  int kc   =  idx >> 13;
  int kk   = ((lane >> 4) << 4) + e;
  int n    = lane & 15;
  int oc   = ot * 16 + n;
  int k    = kc * 32 + kk;
  int tap  = k >> 8;
  int ic   = k & 255;
  wfrag[idx] = f2bf(w[oc * 2304 + ic * 9 + tap]);
}

// NCHW fp32 -> NHWC bf16
__global__ void to_nhwc_bf16(const float* __restrict__ in, unsigned short* __restrict__ out,
                             int H, int W) {
  int idx = blockIdx.x * blockDim.x + threadIdx.x;
  int npix = H * W;
  int tot = NIMG * CC * npix;
  if (idx >= tot) return;
  int x  = idx % W;
  int y  = (idx / W) % H;
  int c  = (idx / npix) % CC;
  int nn = idx / (npix * CC);
  out[((size_t)nn * npix + (size_t)y * W + x) * CC + c] = f2bf(in[idx]);
}

// ---------------------------------------------------------------------------
// Implicit-GEMM 3x3 conv, bf16 WMMA. Block = 128 thr (4 waves).
// Block tile: 32 pixels (M, 2 fragments) x 256 output channels
// (4 waves x 4 WMMA N-tiles) -> 8 WMMAs per wave per K-step.
// K = 9*256 = 2304 in chunks of 32 (72 steps), double-buffered LDS A tile,
// next-step A/B global loads issued before current WMMAs (software pipeline).
// ---------------------------------------------------------------------------
__global__ __launch_bounds__(128) void conv3x3_wmma(
    const unsigned short* __restrict__ xin,    // [N][H*W][256] bf16 NHWC
    const unsigned short* __restrict__ wfrag,  // packed B fragments
    const float* __restrict__ conv_b,
    unsigned short* __restrict__ hout,         // [N][H*W][256] bf16, relu'd
    int H, int W) {
  __shared__ __align__(16) unsigned short a_lds[2 * 32 * 32];  // 2 x 2KB
  int npix = H * W;
  int tile = blockIdx.x;
  int img  = blockIdx.y;
  const unsigned short* xi = xin + (size_t)img * npix * CC;
  unsigned short*       ho = hout + (size_t)img * npix * CC;

  int tid  = threadIdx.x;
  int lane = tid & 31;
  int wv   = tid >> 5;

  // A-tile loader coords (fixed per thread): pixel lm, 8 channels at lcq
  int lm  = tid >> 2;            // 0..31
  int lcq = (tid & 3) * 8;       // 0,8,16,24
  int lp  = tile * 32 + lm;
  int ly = 0, lx = 0;
  bool lvalid = lp < npix;
  if (lvalid) { ly = lp / W; lx = lp % W; }

  // A-fragment coords: row mr, element s holds k = 16*kh + s (contiguous)
  int mr = lane & 15;
  int kh = (lane >> 4) & 1;

  union Frag { u32x4 q[2]; v16bf v; };
  v8f acc0[4], acc1[4];
#pragma unroll
  for (int j = 0; j < 4; ++j) {
    acc0[j] = (v8f){0.f,0.f,0.f,0.f,0.f,0.f,0.f,0.f};
    acc1[j] = (v8f){0.f,0.f,0.f,0.f,0.f,0.f,0.f,0.f};
  }

  auto loadA = [&](int kc, u32x4& gv) {
    int tap = kc >> 3;                       // 8 chunks of 32 per tap
    int dy = tap / 3 - 1, dx = tap % 3 - 1;
    int ic0 = (kc & 7) * 32;
    u32x4 z = {0u, 0u, 0u, 0u};
    gv = z;
    if (lvalid) {
      int iy = ly + dy, ix = lx + dx;
      if (iy >= 0 && iy < H && ix >= 0 && ix < W)
        gv = *(const u32x4*)(xi + ((size_t)iy * W + ix) * CC + ic0 + lcq);
    }
  };
  auto loadB = [&](int kc, Frag* B) {
    const u32x4* bb = (const u32x4*)wfrag +
                      ((size_t)(kc * 16 + wv * 4) * 32 + lane) * 2;
#pragma unroll
    for (int j = 0; j < 4; ++j) { B[j].q[0] = bb[j * 64]; B[j].q[1] = bb[j * 64 + 1]; }
  };

  auto step = [&](int kc, int buf, Frag* Bcur, Frag* Bnxt, u32x4& gv, u32x4& gvn) {
    *(u32x4*)(a_lds + buf * 1024 + lm * 32 + lcq) = gv;
    if (kc + 1 < 72) loadA(kc + 1, gvn);         // next A tile (global, early)
    __syncthreads();
    Frag A0, A1;
    const unsigned short* ab = a_lds + buf * 1024 + mr * 32 + kh * 16;
    A0.q[0] = *(const u32x4*)(ab);
    A0.q[1] = *(const u32x4*)(ab + 8);
    A1.q[0] = *(const u32x4*)(ab + 512);         // pixels 16..31
    A1.q[1] = *(const u32x4*)(ab + 512 + 8);
    if (kc + 1 < 72) loadB(kc + 1, Bnxt);        // next B frags (global, early)
    if (kc + 2 < 72) {                           // prefetch B two steps ahead
      const void* pf = (const void*)((const u32x4*)wfrag +
                       ((size_t)((kc + 2) * 16 + wv * 4) * 32 + lane) * 2);
      __builtin_prefetch(pf, 0, 1);
    }
#pragma unroll
    for (int j = 0; j < 4; ++j) {
      acc0[j] = __builtin_amdgcn_wmma_f32_16x16x32_bf16(
          false, A0.v, false, Bcur[j].v, (short)0, acc0[j], false, false);
      acc1[j] = __builtin_amdgcn_wmma_f32_16x16x32_bf16(
          false, A1.v, false, Bcur[j].v, (short)0, acc1[j], false, false);
    }
  };

  u32x4 gva, gvb;
  Frag Ba[4], Bb[4];
  loadA(0, gva);
  loadB(0, Ba);
  for (int kc = 0; kc < 72; kc += 2) {
    step(kc,     0, Ba, Bb, gva, gvb);
    step(kc + 1, 1, Bb, Ba, gvb, gva);
  }

  // bias + relu + store (C/D layout: vgpr v -> M = v + 8*kh, N = lane&15)
#pragma unroll
  for (int j = 0; j < 4; ++j) {
    int oc = wv * 64 + j * 16 + (lane & 15);
    float cb = conv_b[oc];
#pragma unroll
    for (int v = 0; v < 8; ++v) {
      int p0 = tile * 32 + v + kh * 8;
      if (p0 < npix) {
        float f = acc0[j][v] + cb;
        ho[(size_t)p0 * CC + oc] = f2bf(f > 0.f ? f : 0.f);
      }
      int p1 = tile * 32 + 16 + v + kh * 8;
      if (p1 < npix) {
        float f = acc1[j][v] + cb;
        ho[(size_t)p1 * CC + oc] = f2bf(f > 0.f ? f : 0.f);
      }
    }
  }
}

// ---------------------------------------------------------------------------
// 1x1 heads: 15 outputs (3 logits + 12 reg) per pixel. Wave per pixel.
// ---------------------------------------------------------------------------
__global__ __launch_bounds__(256) void head_kernel(
    const unsigned short* __restrict__ h, const float* __restrict__ logw,
    const float* __restrict__ logb, const float* __restrict__ regw,
    const float* __restrict__ regb, float* __restrict__ logits,
    float* __restrict__ regout, int npix, int aoff) {
  __shared__ float wsm[15 * 256];
  __shared__ float bsm[15];
  for (int i = threadIdx.x; i < 3 * 256; i += 256)  wsm[i]       = logw[i];
  for (int i = threadIdx.x; i < 12 * 256; i += 256) wsm[768 + i] = regw[i];
  if (threadIdx.x < 3)       bsm[threadIdx.x] = logb[threadIdx.x];
  else if (threadIdx.x < 15) bsm[threadIdx.x] = regb[threadIdx.x - 3];
  __syncthreads();

  int wv = threadIdx.x >> 5, lane = threadIdx.x & 31;
  int pix = blockIdx.x * 8 + wv;
  int img = blockIdx.y;
  if (pix >= npix) return;
  const unsigned short* hp = h + ((size_t)img * npix + pix) * CC;

  float s[15];
#pragma unroll
  for (int o = 0; o < 15; ++o) s[o] = 0.f;
  for (int it = 0; it < 8; ++it) {
    int ch = lane + it * 32;
    float xv = bf2f(hp[ch]);
#pragma unroll
    for (int o = 0; o < 15; ++o) s[o] += xv * wsm[o * 256 + ch];
  }
#pragma unroll
  for (int o = 0; o < 15; ++o)
    for (int off = 16; off > 0; off >>= 1)
      s[o] += __shfl_down(s[o], off, 32);

  if (lane == 0) {
    int ab = aoff + pix * 3;
    float* lg = logits + (size_t)img * TOTA;
    float* rg = regout + (size_t)img * TOTA * 4;
    for (int o = 0; o < 3; ++o)  lg[ab + o] = s[o] + bsm[o];
    for (int o = 0; o < 12; ++o) rg[(size_t)(ab + (o >> 2)) * 4 + (o & 3)] = s[3 + o] + bsm[3 + o];
  }
}

// ---------------------------------------------------------------------------
// Per (image, level) iterative top-k via block argmax (destroys logits).
// ---------------------------------------------------------------------------
__global__ __launch_bounds__(1024) void topk_kernel(float* __restrict__ logits,
    float* __restrict__ topVal, int* __restrict__ topIdx, LvlInfo L) {
  __shared__ float rv[1024];
  __shared__ int   ri[1024];
  int lvl = blockIdx.x, img = blockIdx.y;
  float* lg = logits + (size_t)img * TOTA + L.aoff[lvl];
  int nA = L.nA[lvl], kc = L.kcnt[lvl];
  int tid = threadIdx.x;
  for (int k = 0; k < kc; ++k) {
    float bv = NEGINF; int bi = 0x7fffffff;
    for (int i = tid; i < nA; i += 1024) {
      float v = lg[i];
      if (v > bv || (v == bv && i < bi)) { bv = v; bi = i; }
    }
    rv[tid] = bv; ri[tid] = bi;
    __syncthreads();
    for (int s2 = 512; s2 > 0; s2 >>= 1) {
      if (tid < s2) {
        float ov = rv[tid + s2]; int oi = ri[tid + s2];
        if (ov > rv[tid] || (ov == rv[tid] && oi < ri[tid])) { rv[tid] = ov; ri[tid] = oi; }
      }
      __syncthreads();
    }
    if (tid == 0) {
      int slot = img * KSEL + L.koff[lvl] + k;
      topVal[slot] = rv[0];
      topIdx[slot] = ri[0];
      lg[ri[0]] = NEGINF;
    }
    __syncthreads();
  }
}

__global__ void init_cm(float* cm) { *cm = 0.f; }

// ---------------------------------------------------------------------------
// Box decode + clamp + sigmoid + group id + global coord max.
// ---------------------------------------------------------------------------
__global__ void decode_kernel(const float* __restrict__ topVal, const int* __restrict__ topIdx,
    const float* p0, const float* p1, const float* p2, const float* p3, const float* p4,
    const float* __restrict__ reg, const float* __restrict__ imsz,
    float* __restrict__ boxes, float* __restrict__ score, int* __restrict__ imidx,
    int* __restrict__ group, float* cm, LvlInfo L) {
  int i = blockIdx.x * blockDim.x + threadIdx.x;
  if (i >= NIMG * KSEL) return;
  const float* pr[NLVL] = {p0, p1, p2, p3, p4};
  int img = i / KSEL, slot = i - img * KSEL;
  int lvl = 0;
  for (int l = 1; l < NLVL; ++l) if (slot >= L.koff[l]) lvl = l;
  int top = topIdx[i];
  float val = topVal[i];
  const float* p = pr[lvl] + (size_t)top * 4;
  const float* r = reg + ((size_t)img * TOTA + L.aoff[lvl] + top) * 4;
  float pw = p[2] - p[0], ph = p[3] - p[1];
  float pcx = p[0] + 0.5f * pw, pcy = p[1] + 0.5f * ph;
  float cx = pcx + r[0] * pw, cy = pcy + r[1] * ph;
  float bw = pw * __expf(r[2]), bh = ph * __expf(r[3]);
  float x1 = cx - 0.5f * bw, y1 = cy - 0.5f * bh;
  float x2 = cx + 0.5f * bw, y2 = cy + 0.5f * bh;
  float imh = imsz[img * 2], imw = imsz[img * 2 + 1];
  x1 = fminf(fmaxf(x1, 0.f), imw); x2 = fminf(fmaxf(x2, 0.f), imw);
  y1 = fminf(fmaxf(y1, 0.f), imh); y2 = fminf(fmaxf(y2, 0.f), imh);
  boxes[i * 4 + 0] = x1; boxes[i * 4 + 1] = y1;
  boxes[i * 4 + 2] = x2; boxes[i * 4 + 3] = y2;
  score[i] = 1.f / (1.f + __expf(-val));
  imidx[i] = img;
  group[i] = img * 10 + lvl;
  float mx = fmaxf(fmaxf(x1, y1), fmaxf(x2, y2));
  atomicMax((int*)cm, __float_as_int(mx));    // coords >= 0 after clamp
}

__global__ void sortprep_kernel(const float* __restrict__ boxes, const float* __restrict__ score,
    const int* __restrict__ group, const float* __restrict__ cm,
    float* __restrict__ keys, int* __restrict__ idxs, float* __restrict__ boxesOff) {
  int j = blockIdx.x * blockDim.x + threadIdx.x;
  if (j >= NP2) return;
  const int M = NIMG * KSEL;
  if (j < M) {
    keys[j] = score[j];
    idxs[j] = j;
    float off = (float)group[j] * (*cm + 1.0f);
    for (int c = 0; c < 4; ++c) boxesOff[j * 4 + c] = boxes[j * 4 + c] + off;
  } else { keys[j] = NEGINF; idxs[j] = -1; }
}

// single-block bitonic sort, descending by key (stable via index tie-break)
__global__ __launch_bounds__(1024) void sort_kernel(float* __restrict__ keys, int* __restrict__ idxs) {
  int tid = threadIdx.x;
  for (unsigned k = 2; k <= NP2; k <<= 1) {
    for (unsigned j = k >> 1; j > 0; j >>= 1) {
      for (unsigned i = tid; i < NP2; i += 1024) {
        unsigned l = i ^ j;
        if (l > i) {
          float a = keys[i], b = keys[l];
          int  ia = idxs[i], ib = idxs[l];
          bool up = ((i & k) == 0);
          bool sw = up ? (a < b || (a == b && ia > ib))
                       : (a > b || (a == b && ia < ib));
          if (sw) { keys[i] = b; keys[l] = a; idxs[i] = ib; idxs[l] = ia; }
        }
      }
      __syncthreads();
    }
  }
}

__global__ void gather_kernel(const float* __restrict__ keys, const int* __restrict__ idxs,
    const float* __restrict__ boxes, const float* __restrict__ boxesOff,
    const int* __restrict__ imidx, float* __restrict__ bs, float* __restrict__ braw,
    float* __restrict__ ssS, int* __restrict__ imsS, float* __restrict__ areaS) {
  int j = blockIdx.x * blockDim.x + threadIdx.x;
  if (j >= NIMG * KSEL) return;
  int o = idxs[j];
  ssS[j] = keys[j];
  float b0 = boxesOff[o * 4], b1 = boxesOff[o * 4 + 1];
  float b2 = boxesOff[o * 4 + 2], b3 = boxesOff[o * 4 + 3];
  bs[j * 4] = b0; bs[j * 4 + 1] = b1; bs[j * 4 + 2] = b2; bs[j * 4 + 3] = b3;
  for (int c = 0; c < 4; ++c) braw[j * 4 + c] = boxes[o * 4 + c];
  imsS[j] = imidx[o];
  areaS[j] = (b2 - b0) * (b3 - b1);
}

__global__ __launch_bounds__(1024) void nms_kernel(const float* __restrict__ bs,
    const float* __restrict__ areaS, unsigned char* __restrict__ keepG) {
  __shared__ unsigned char keep[NIMG * KSEL];
  const int M = NIMG * KSEL;
  int tid = threadIdx.x;
  for (int j = tid; j < M; j += 1024) keep[j] = 1;
  __syncthreads();
  for (int i = 0; i < M; ++i) {
    if (keep[i]) {
      float x1 = bs[i * 4], y1 = bs[i * 4 + 1], x2 = bs[i * 4 + 2], y2 = bs[i * 4 + 3];
      float ai = areaS[i];
      for (int j2 = i + 1 + tid; j2 < M; j2 += 1024) {
        if (!keep[j2]) continue;
        float ix1 = fmaxf(x1, bs[j2 * 4]);
        float iy1 = fmaxf(y1, bs[j2 * 4 + 1]);
        float ix2 = fminf(x2, bs[j2 * 4 + 2]);
        float iy2 = fminf(y2, bs[j2 * 4 + 3]);
        float inter = fmaxf(ix2 - ix1, 0.f) * fmaxf(iy2 - iy1, 0.f);
        float iou = inter / fmaxf(ai + areaS[j2] - inter, 1e-9f);
        if (iou > 0.7f) keep[j2] = 0;
      }
    }
    __syncthreads();
  }
  for (int j = tid; j < M; j += 1024) keepG[j] = keep[j];
}

__global__ void select_kernel(const float* __restrict__ braw, const int* __restrict__ imsS,
    const unsigned char* __restrict__ keepG, float* __restrict__ outB, int* __restrict__ outI) {
  int img = blockIdx.x;
  if (threadIdx.x != 0) return;
  const int M = NIMG * KSEL;
  int cnt = 0;
  for (int j = 0; j < M && cnt < 1000; ++j) {
    if (keepG[j] && imsS[j] == img) {
      for (int c = 0; c < 4; ++c) outB[(size_t)(img * 1000 + cnt) * 4 + c] = braw[j * 4 + c];
      outI[img * 1000 + cnt] = img;
      ++cnt;
    }
  }
  for (; cnt < 1000; ++cnt) {
    for (int c = 0; c < 4; ++c) outB[(size_t)(img * 1000 + cnt) * 4 + c] = 0.f;
    outI[img * 1000 + cnt] = -1;
  }
}

// ---------------------------------------------------------------------------
extern "C" void kernel_launch(void* const* d_in, const int* in_sizes, int n_in,
                              void* d_out, int out_size, void* d_ws, size_t ws_size,
                              hipStream_t stream) {
  const float* fmap[NLVL]   = {(const float*)d_in[0], (const float*)d_in[2],
                               (const float*)d_in[4], (const float*)d_in[6],
                               (const float*)d_in[8]};
  const float* priors[NLVL] = {(const float*)d_in[1], (const float*)d_in[3],
                               (const float*)d_in[5], (const float*)d_in[7],
                               (const float*)d_in[9]};
  const float* imsz   = (const float*)d_in[10];
  const float* conv_w = (const float*)d_in[11];
  const float* conv_b = (const float*)d_in[12];
  const float* log_w  = (const float*)d_in[13];
  const float* log_b  = (const float*)d_in[14];
  const float* reg_w  = (const float*)d_in[15];
  const float* reg_b  = (const float*)d_in[16];

  static const int Hs[NLVL] = {160, 80, 40, 20, 10};
  static const int Ws[NLVL] = {240, 120, 60, 30, 15};

  LvlInfo L;
  { int ao = 0, ko = 0;
    for (int l = 0; l < NLVL; ++l) {
      int nA = Hs[l] * Ws[l] * 3;
      L.aoff[l] = ao; L.nA[l] = nA; ao += nA;
      int kc = nA < 1000 ? nA : 1000;
      L.koff[l] = ko; L.kcnt[l] = kc; ko += kc;
    } }

  const int M = NIMG * KSEL;
  char* base = (char*)d_ws;
  size_t cur = 0;
  auto alloc = [&](size_t bytes) -> char* {
    char* p = base + cur;
    cur = (cur + bytes + 255) & ~(size_t)255;
    return p;
  };
  unsigned short* wfrag   = (unsigned short*)alloc((size_t)72 * 16 * 32 * 16 * 2);
  unsigned short* xnhwc   = (unsigned short*)alloc((size_t)NIMG * 38400 * CC * 2); // max level
  unsigned short* hbuf    = (unsigned short*)alloc((size_t)NIMG * 38400 * CC * 2);
  float*          logits  = (float*)alloc((size_t)NIMG * TOTA * 4);
  float*          regbuf  = (float*)alloc((size_t)NIMG * TOTA * 4 * 4);
  float*          topVal  = (float*)alloc((size_t)M * 4);
  int*            topIdx  = (int*)alloc((size_t)M * 4);
  float*          boxes   = (float*)alloc((size_t)M * 4 * 4);
  float*          score   = (float*)alloc((size_t)M * 4);
  int*            imidx   = (int*)alloc((size_t)M * 4);
  int*            group   = (int*)alloc((size_t)M * 4);
  float*          boxesOff= (float*)alloc((size_t)M * 4 * 4);
  float*          keys    = (float*)alloc((size_t)NP2 * 4);
  int*            idxs    = (int*)alloc((size_t)NP2 * 4);
  float*          bs      = (float*)alloc((size_t)M * 4 * 4);
  float*          braw    = (float*)alloc((size_t)M * 4 * 4);
  float*          ssS     = (float*)alloc((size_t)M * 4);
  int*            imsS    = (int*)alloc((size_t)M * 4);
  float*          areaS   = (float*)alloc((size_t)M * 4);
  unsigned char*  keepG   = (unsigned char*)alloc((size_t)M);
  float*          cm      = (float*)alloc(256);
  (void)ws_size; (void)in_sizes; (void)n_in; (void)out_size;

  init_cm<<<1, 1, 0, stream>>>(cm);
  pack_weights<<<(72 * 16 * 32 * 16 + 255) / 256, 256, 0, stream>>>(conv_w, wfrag);

  for (int l = 0; l < NLVL; ++l) {
    int H = Hs[l], W = Ws[l], npix = H * W;
    int tot = NIMG * CC * npix;
    to_nhwc_bf16<<<(tot + 255) / 256, 256, 0, stream>>>(fmap[l], xnhwc, H, W);
    conv3x3_wmma<<<dim3((npix + 31) / 32, NIMG), 128, 0, stream>>>(
        xnhwc, wfrag, conv_b, hbuf, H, W);
    head_kernel<<<dim3((npix + 7) / 8, NIMG), 256, 0, stream>>>(
        hbuf, log_w, log_b, reg_w, reg_b, logits, regbuf, npix, L.aoff[l]);
  }

  topk_kernel<<<dim3(NLVL, NIMG), 1024, 0, stream>>>(logits, topVal, topIdx, L);
  decode_kernel<<<(M + 255) / 256, 256, 0, stream>>>(
      topVal, topIdx, priors[0], priors[1], priors[2], priors[3], priors[4],
      regbuf, imsz, boxes, score, imidx, group, cm, L);
  sortprep_kernel<<<(NP2 + 255) / 256, 256, 0, stream>>>(boxes, score, group, cm,
                                                         keys, idxs, boxesOff);
  sort_kernel<<<1, 1024, 0, stream>>>(keys, idxs);
  gather_kernel<<<(M + 255) / 256, 256, 0, stream>>>(keys, idxs, boxes, boxesOff,
                                                     imidx, bs, braw, ssS, imsS, areaS);
  nms_kernel<<<1, 1024, 0, stream>>>(bs, areaS, keepG);

  float* outB = (float*)d_out;
  int*   outI = (int*)((float*)d_out + (size_t)NIMG * 1000 * 4);
  select_kernel<<<NIMG, 64, 0, stream>>>(braw, imsS, keepG, outB, outI);
}
